// DPRNNBase_51221779972627
// MI455X (gfx1250) — matrix operand order repaired and checked by reading, not compiled, not run on hardware
//
#include <hip/hip_runtime.h>

#define T_SEQ 2048
#define BATCH 32
#define IN0   256
#define HID   256
#define G4    1024            // 4*H
#define MROWS (T_SEQ * BATCH) // 65536

typedef __attribute__((ext_vector_type(16))) _Float16 v16h;
typedef __attribute__((ext_vector_type(8)))  float    v8f;
typedef __attribute__((ext_vector_type(4)))  int      v4i;

#if defined(__gfx1250__) && __has_builtin(__builtin_amdgcn_global_load_async_to_lds_b128)
#define USE_ASYNC_LDS 1
#else
#define USE_ASYNC_LDS 0
#endif

#if USE_ASYNC_LDS
typedef __attribute__((address_space(1))) v4i* gvec_t;
typedef __attribute__((address_space(3))) v4i* svec_t;
__device__ __forceinline__ void async_cp16(const _Float16* gsrc, _Float16* ldst) {
    __builtin_amdgcn_global_load_async_to_lds_b128(
        (gvec_t)(void*)const_cast<_Float16*>(gsrc), (svec_t)(void*)ldst, 0, 0);
}
__device__ __forceinline__ void wait_async0() {
#if __has_builtin(__builtin_amdgcn_s_wait_asynccnt)
    __builtin_amdgcn_s_wait_asynccnt(0);
#else
    asm volatile("s_wait_asynccnt 0" ::: "memory");
#endif
}
#endif

union Frag {
    v16h v;
    uint4 q[2];
};

// A-operand (16x32 f16, MxK): lane L holds row M=L%16;
// lanes 0-15 hold K 0..15, lanes 16-31 hold K 16..31,
// packed 2 f16 per dword -> dwords {kb..kb+7}, kb = (L<16?0:8).
__device__ __forceinline__ v16h load_a_frag(const uint32_t* tile, int row0,
                                            int strideDw, int kcDw, int lane) {
    int r  = row0 + (lane & 15);
    int kb = (lane < 16) ? 0 : 8;
    const uint32_t* p = tile + (size_t)r * strideDw + kcDw + kb;
    Frag f;
    f.q[0] = *(const uint4*)p;
    f.q[1] = *(const uint4*)(p + 4);
    return f.v;
}

// B-operand (32x16 f16, KxN): lane L holds column N=L%16; lanes 0-15 hold
// K 0..15, lanes 16-31 hold K 16..31 -> 8 consecutive dwords of the
// W-row-major tile row.
__device__ __forceinline__ v16h load_b_frag(const uint32_t* tile, int col0,
                                            int strideDw, int kcDw, int lane) {
    int c  = col0 + (lane & 15);
    int kb = (lane < 16) ? 0 : 8;
    const uint32_t* p = tile + (size_t)c * strideDw + kcDw + kb;
    Frag f;
    f.q[0] = *(const uint4*)p;
    f.q[1] = *(const uint4*)(p + 4);
    return f.v;
}

__device__ __forceinline__ float sigmoid_f(float x) {
    return 1.0f / (1.0f + __expf(-x));
}
__device__ __forceinline__ float tanh_f(float x) {
    return 1.0f - 2.0f / (__expf(2.0f * x) + 1.0f);
}

__global__ void cvt_f32_f16(const float* __restrict__ s, _Float16* __restrict__ d, int n) {
    int i = blockIdx.x * blockDim.x + threadIdx.x;
    int stride = gridDim.x * blockDim.x;
    for (; i < n; i += stride) d[i] = (_Float16)s[i];
}

__global__ void add_bias(const float* __restrict__ a, const float* __restrict__ b,
                         float* __restrict__ o, int n) {
    int i = blockIdx.x * blockDim.x + threadIdx.x;
    if (i < n) o[i] = a[i] + b[i];
}

// xg[dir] = X_f16 @ W_ih[dir]^T + (b_ih + b_hh)[dir]   (result stored f16)
// X: [MROWS, K] f16 row-major. W: [2, G4, K] f16 row-major. out: [2, MROWS, G4] f16.
// Block tile 64(M) x 128(N), 8 waves, each wave 2x2 WMMA tiles.
// Double-buffered LDS staging via GLOBAL_LOAD_ASYNC_TO_LDS (ASYNCcnt) when available.
__global__ __launch_bounds__(256) void gemm_ih(
    const _Float16* __restrict__ X,
    const _Float16* __restrict__ W,
    const float*    __restrict__ bias,
    _Float16*       __restrict__ out,
    int K)
{
    __shared__ _Float16 xs[2][64 * 32];
    __shared__ _Float16 wsm[2][128 * 32];

    const int tid  = threadIdx.x;
    const int lane = tid & 31;
    const int wv   = tid >> 5;
    const int mb   = blockIdx.x * 64;
    const int nb   = blockIdx.y * 128;
    const int dir  = blockIdx.z;
    const _Float16* Wd = W + (size_t)dir * G4 * K;

    const int m0 = (wv >> 2) * 32; // wave grid 2(M) x 4(N)
    const int n0 = (wv & 3) * 32;

    // Per-thread staging addresses (16B X, 32B W per thread per k-step)
    const int xrow = tid >> 2, xcg = tid & 3;
    const int wrow = tid >> 1, wcg = tid & 1;

    auto stage = [&](int kc, int buf) {
        const _Float16* xsrc = X  + (size_t)(mb + xrow) * K + kc + xcg * 8;
        const _Float16* wsrc = Wd + (size_t)(nb + wrow) * K + kc + wcg * 16;
        _Float16* xdst = &xs[buf][xrow * 32 + xcg * 8];
        _Float16* wdst = &wsm[buf][wrow * 32 + wcg * 16];
#if USE_ASYNC_LDS
        async_cp16(xsrc, xdst);
        async_cp16(wsrc, wdst);
        async_cp16(wsrc + 8, wdst + 8);
#else
        *(uint4*)xdst       = *(const uint4*)xsrc;
        *(uint4*)wdst       = *(const uint4*)wsrc;
        *(uint4*)(wdst + 8) = *(const uint4*)(wsrc + 8);
#endif
    };

    v8f acc[2][2];
#pragma unroll
    for (int i = 0; i < 2; i++)
#pragma unroll
        for (int j = 0; j < 2; j++) acc[i][j] = (v8f){};

    const int nkc = K / 32;
    stage(0, 0);
    int buf = 0;
    for (int ki = 0; ki < nkc; ki++) {
#if USE_ASYNC_LDS
        wait_async0();           // this thread's staged tile has landed in LDS
#endif
        __syncthreads();         // everyone's tile data visible; prev buffer free
        if (ki + 1 < nkc) stage((ki + 1) * 32, buf ^ 1);

        const uint32_t* xsd = (const uint32_t*)xs[buf];
        const uint32_t* wsd = (const uint32_t*)wsm[buf];
        v16h a[2], b[2];
#pragma unroll
        for (int i = 0; i < 2; i++) a[i] = load_a_frag(xsd, m0 + i * 16, 16, 0, lane);
#pragma unroll
        for (int j = 0; j < 2; j++) b[j] = load_b_frag(wsd, n0 + j * 16, 16, 0, lane);
#pragma unroll
        for (int i = 0; i < 2; i++)
#pragma unroll
            for (int j = 0; j < 2; j++)
                acc[i][j] = __builtin_amdgcn_wmma_f32_16x16x32_f16(
                    false, a[i], false, b[j], (short)0, acc[i][j], false, false);
        buf ^= 1;
    }

    // Epilogue: C layout -> lane holds col N=L%16, VGPR r holds row r + 8*(L>=16)
    const int nl = lane & 15;
    const int mo = (lane < 16) ? 0 : 8;
    _Float16* obase = out + (size_t)dir * MROWS * G4;
#pragma unroll
    for (int i = 0; i < 2; i++) {
#pragma unroll
        for (int j = 0; j < 2; j++) {
            int n = nb + n0 + j * 16 + nl;
            float bv = bias[dir * G4 + n];
#pragma unroll
            for (int r = 0; r < 8; r++) {
                int m = mb + m0 + i * 16 + r + mo;
                obase[(size_t)m * G4 + n] = (_Float16)(acc[i][j][r] + bv);
            }
        }
    }
}

// Persistent recurrent scan: 1 workgroup per direction, 16 waves.
// W_hh held entirely in VGPRs as WMMA B-fragments (4 Ntiles x 8 Kchunks/wave
// = 256 VGPRs), zero per-step weight traffic. h (f16) + gates (f32) in the
// 320KB WGP LDS; cell state c in registers. Next step's xg row prefetched.
__global__ __launch_bounds__(512) void lstm_scan(
    const _Float16* __restrict__ xg,   // [2, MROWS, G4] f16
    const _Float16* __restrict__ whh,  // [2, G4, HID] f16
    _Float16*       __restrict__ out_h,// [T,B,512] f16 slot dir*256 (or null)
    float*          __restrict__ out_f)// [T,B,512] f32 slot dir*256 (or null)
{
    extern __shared__ char smem[];
    _Float16* h_lds = (_Float16*)smem;                  // 32*256 f16 = 16KB
    float*    gates = (float*)(smem + 32 * 256 * 2);    // 32*1024 f32 = 128KB

    const int tid  = threadIdx.x;
    const int lane = tid & 31;
    const int wv   = tid >> 5;      // 0..15, owns N cols [wv*64, wv*64+64)
    const int dir  = blockIdx.x;

    for (int i = tid; i < 32 * 256; i += 512) h_lds[i] = (_Float16)0.0f;

    // Preload W_hh fragments (resident in VGPRs for all 2048 steps)
    v16h bfr[4][8];
    const uint32_t* whd = (const uint32_t*)whh;
#pragma unroll
    for (int nt = 0; nt < 4; nt++) {
        int n = (wv * 4 + nt) * 16 + (lane & 15);
        const uint32_t* rowp = whd + ((size_t)dir * G4 + n) * (HID / 2);
#pragma unroll
        for (int kc = 0; kc < 8; kc++) {
            const uint32_t* p = rowp + kc * 16 + ((lane < 16) ? 0 : 8);
            Frag f;
            f.q[0] = *(const uint4*)p;
            f.q[1] = *(const uint4*)(p + 4);
            bfr[nt][kc] = f.v;
        }
    }

    float cst[16];
#pragma unroll
    for (int j = 0; j < 16; j++) cst[j] = 0.0f;

    __syncthreads();

    const int b2 = tid & 31;         // phase-2 batch row
    const int jb = (tid >> 5) * 16;  // phase-2 hidden-col base

    for (int step = 0; step < T_SEQ; step++) {
        const int tt = dir ? (T_SEQ - 1 - step) : step;
        const _Float16* xrow = xg + ((size_t)dir * MROWS + (size_t)tt * BATCH) * G4;

        // Prefetch next step's xg row (64KB) while this step computes
        if (step + 1 < T_SEQ) {
            const int tn = dir ? (T_SEQ - 2 - step) : (step + 1);
            const char* pf = (const char*)(xg + ((size_t)dir * MROWS + (size_t)tn * BATCH) * G4);
            __builtin_prefetch(pf + tid * 128, 0, 1);
        }

        const uint32_t* hld = (const uint32_t*)h_lds;
        // Phase 1: gates = h @ W_hh^T + xg[t]   (128 WMMA tiles across 16 waves)
#pragma unroll
        for (int mt = 0; mt < 2; mt++) {
            v16h a[8];
#pragma unroll
            for (int kc = 0; kc < 8; kc++)
                a[kc] = load_a_frag(hld, mt * 16, HID / 2, kc * 16, lane);
#pragma unroll
            for (int nt = 0; nt < 4; nt++) {
                v8f acc = (v8f){};
#pragma unroll
                for (int kc = 0; kc < 8; kc++)
                    acc = __builtin_amdgcn_wmma_f32_16x16x32_f16(
                        false, a[kc], false, bfr[nt][kc], (short)0, acc, false, false);
                int n  = (wv * 4 + nt) * 16 + (lane & 15);
                int mo = (lane < 16) ? 0 : 8;
#pragma unroll
                for (int r = 0; r < 8; r++) {
                    int m = mt * 16 + r + mo;
                    gates[m * G4 + n] = acc[r] + (float)xrow[(size_t)m * G4 + n];
                }
            }
        }
        __syncthreads();

        // Phase 2: LSTM cell update (i,f,g,o split along N), c in registers
#pragma unroll
        for (int jj = 0; jj < 16; jj++) {
            int j = jb + jj;
            float gi = gates[b2 * G4 + j];
            float gf = gates[b2 * G4 + j + 256];
            float gg = gates[b2 * G4 + j + 512];
            float go = gates[b2 * G4 + j + 768];
            float c = sigmoid_f(gf) * cst[jj] + sigmoid_f(gi) * tanh_f(gg);
            cst[jj] = c;
            float h = sigmoid_f(go) * tanh_f(c);
            h_lds[b2 * 256 + j] = (_Float16)h;
            size_t oidx = ((size_t)tt * BATCH + b2) * 512 + (size_t)dir * 256 + j;
            if (out_h) out_h[oidx] = (_Float16)h;
            if (out_f) out_f[oidx] = h;
        }
        __syncthreads();
    }
}

extern "C" void kernel_launch(void* const* d_in, const int* in_sizes, int n_in,
                              void* d_out, int out_size, void* d_ws, size_t ws_size,
                              hipStream_t stream) {
    (void)in_sizes; (void)n_in; (void)out_size; (void)ws_size;

    const float* x    = (const float*)d_in[0];
    const float* wih0 = (const float*)d_in[1];
    const float* whh0 = (const float*)d_in[2];
    const float* bih0 = (const float*)d_in[3];
    const float* bhh0 = (const float*)d_in[4];
    const float* wih1 = (const float*)d_in[5];
    const float* whh1 = (const float*)d_in[6];
    const float* bih1 = (const float*)d_in[7];
    const float* bhh1 = (const float*)d_in[8];

    char* ws = (char*)d_ws;
    size_t off = 0;
    auto carve = [&](size_t bytes) -> char* {
        char* p = ws + off;
        off += (bytes + 255) & ~(size_t)255;
        return p;
    };
    _Float16* xbf   = (_Float16*)carve((size_t)MROWS * IN0 * 2);
    _Float16* wih0h = (_Float16*)carve((size_t)2 * G4 * IN0 * 2);
    _Float16* whh0h = (_Float16*)carve((size_t)2 * G4 * HID * 2);
    _Float16* wih1h = (_Float16*)carve((size_t)2 * G4 * 512 * 2);
    _Float16* whh1h = (_Float16*)carve((size_t)2 * G4 * HID * 2);
    float*    bias0 = (float*)carve((size_t)2 * G4 * 4);
    float*    bias1 = (float*)carve((size_t)2 * G4 * 4);
    _Float16* xgbuf = (_Float16*)carve((size_t)2 * MROWS * G4 * 2);
    _Float16* out0h = (_Float16*)carve((size_t)MROWS * 512 * 2);

    cvt_f32_f16<<<2048, 256, 0, stream>>>(x, xbf, MROWS * IN0);
    cvt_f32_f16<<<512, 256, 0, stream>>>(wih0, wih0h, 2 * G4 * IN0);
    cvt_f32_f16<<<512, 256, 0, stream>>>(whh0, whh0h, 2 * G4 * HID);
    cvt_f32_f16<<<512, 256, 0, stream>>>(wih1, wih1h, 2 * G4 * 512);
    cvt_f32_f16<<<512, 256, 0, stream>>>(whh1, whh1h, 2 * G4 * HID);
    add_bias<<<8, 256, 0, stream>>>(bih0, bhh0, bias0, 2 * G4);
    add_bias<<<8, 256, 0, stream>>>(bih1, bhh1, bias1, 2 * G4);

    dim3 gg(MROWS / 64, G4 / 128, 2);
    size_t smem = 32 * 256 * 2 + 32 * 1024 * 4; // 147456 B < 320KB WGP LDS
    (void)hipFuncSetAttribute((const void*)lstm_scan,
                              hipFuncAttributeMaxDynamicSharedMemorySize, (int)smem);

    // Layer 0
    gemm_ih<<<gg, 256, 0, stream>>>(xbf, wih0h, bias0, xgbuf, IN0);
    lstm_scan<<<dim3(2), 512, smem, stream>>>(xgbuf, whh0h, out0h, nullptr);
    // Layer 1
    gemm_ih<<<gg, 256, 0, stream>>>(out0h, wih1h, bias1, xgbuf, 512);
    lstm_scan<<<dim3(2), 512, smem, stream>>>(xgbuf, whh1h, nullptr, (float*)d_out);
}
